// IrBinaryLinear_44349832298615
// MI455X (gfx1250) — matrix operand order repaired and checked by reading, not compiled, not run on hardware
//
#include <hip/hip_runtime.h>
#include <hip/hip_bf16.h>

// ---------------------------------------------------------------------------
// out[b,s,o] = scale[o] * sum_i x[b,s,i] * sign(w[o,i] - mean_i(w[o,:])) + bias[o]
// M = 4*2048 = 8192, K = 4096, N = 4096.
// Phase 1: per-row weight stats + f16 sign matrix (B^T layout [N][K]).
// Phase 2: x f32 -> f16.
// Phase 3: f16 WMMA GEMM (v_wmma_f32_16x16x32_f16), 64x64 wave tile,
//          f32 epilogue with scale+bias.
// ---------------------------------------------------------------------------

typedef __attribute__((ext_vector_type(16))) _Float16 v16h;
typedef __attribute__((ext_vector_type(8)))  _Float16 v8h;
typedef __attribute__((ext_vector_type(4)))  _Float16 v4h;
typedef __attribute__((ext_vector_type(8)))  float    v8f;
typedef __attribute__((ext_vector_type(4)))  float    v4f;

#define M_DIM 8192
#define N_DIM 4096
#define K_DIM 4096

// ---------------------------------------------------------------------------
// Kernel 1: per output row o: mean, scale = mean|w-mean|, signs (f16 +/-1).
// ---------------------------------------------------------------------------
__global__ __launch_bounds__(256) void prep_weight_kernel(
    const float* __restrict__ w,
    _Float16* __restrict__ signs,   // [N][K] row-major (this is B^T)
    float* __restrict__ scale)      // [N]
{
    __shared__ float row[K_DIM];    // 16 KB
    __shared__ float red[256];

    const int o   = blockIdx.x;
    const int tid = threadIdx.x;
    const float* wr = w + (size_t)o * K_DIM;

    float s = 0.0f;
    for (int i = tid * 4; i < K_DIM; i += 256 * 4) {
        v4f v = *(const v4f*)(wr + i);
        *(v4f*)(row + i) = v;
        s += v.x + v.y + v.z + v.w;
    }
    red[tid] = s;
    __syncthreads();
    for (int off = 128; off > 0; off >>= 1) {
        if (tid < off) red[tid] += red[tid + off];
        __syncthreads();
    }
    const float mean = red[0] * (1.0f / (float)K_DIM);
    __syncthreads();

    float sa = 0.0f;
    for (int i = tid * 4; i < K_DIM; i += 256 * 4) {
        v4f v = *(const v4f*)(row + i);
        sa += fabsf(v.x - mean) + fabsf(v.y - mean) +
              fabsf(v.z - mean) + fabsf(v.w - mean);
    }
    red[tid] = sa;
    __syncthreads();
    for (int off = 128; off > 0; off >>= 1) {
        if (tid < off) red[tid] += red[tid + off];
        __syncthreads();
    }
    if (tid == 0) scale[o] = red[0] * (1.0f / (float)K_DIM);

    for (int i = tid * 4; i < K_DIM; i += 256 * 4) {
        v4f v = *(const v4f*)(row + i);
        v4h h;
        float d0 = v.x - mean, d1 = v.y - mean, d2 = v.z - mean, d3 = v.w - mean;
        h.x = (_Float16)((d0 > 0.0f) ? 1.0f : ((d0 < 0.0f) ? -1.0f : 0.0f));
        h.y = (_Float16)((d1 > 0.0f) ? 1.0f : ((d1 < 0.0f) ? -1.0f : 0.0f));
        h.z = (_Float16)((d2 > 0.0f) ? 1.0f : ((d2 < 0.0f) ? -1.0f : 0.0f));
        h.w = (_Float16)((d3 > 0.0f) ? 1.0f : ((d3 < 0.0f) ? -1.0f : 0.0f));
        *(v4h*)(signs + (size_t)o * K_DIM + i) = h;
    }
}

// ---------------------------------------------------------------------------
// Kernel 2: x f32 -> f16, vectorized 8 elements/thread (exact coverage).
// ---------------------------------------------------------------------------
__global__ __launch_bounds__(256) void conv_x_kernel(
    const float* __restrict__ x,
    _Float16* __restrict__ xh)
{
    const size_t base = ((size_t)blockIdx.x * 256 + threadIdx.x) * 8;
    v4f a = *(const v4f*)(x + base);
    v4f b = *(const v4f*)(x + base + 4);
    v8h o;
    o[0] = (_Float16)a.x; o[1] = (_Float16)a.y;
    o[2] = (_Float16)a.z; o[3] = (_Float16)a.w;
    o[4] = (_Float16)b.x; o[5] = (_Float16)b.y;
    o[6] = (_Float16)b.z; o[7] = (_Float16)b.w;
    *(v8h*)(xh + base) = o;
}

// ---------------------------------------------------------------------------
// Kernel 3: GEMM. Block = 256 threads = 8 waves (wave32).
// Block tile 128(M) x 256(N); waves 2(M) x 4(N); per-wave 64x64 = 4x4 tiles.
// 8 fragment loads feed 16 WMMAs per K=32 step (512 B L2 traffic / WMMA).
// A = xh [M][K] f16, B = signs [N][K] f16 (B^T layout -> contiguous K loads).
// ---------------------------------------------------------------------------
__global__ __launch_bounds__(256) void gemm_wmma_kernel(
    const _Float16* __restrict__ A,
    const _Float16* __restrict__ B,
    const float* __restrict__ scale,
    const float* __restrict__ bias,
    float* __restrict__ C)
{
    const int lane  = threadIdx.x & 31;
    const int wave  = threadIdx.x >> 5;
    const int waveM = wave & 1;     // 2 waves along M
    const int waveN = wave >> 1;    // 4 waves along N
    const int mBase = blockIdx.y * 128 + waveM * 64;
    const int nBase = blockIdx.x * 256 + waveN * 64;
    const int l15   = lane & 15;
    const int hi    = lane >> 4;    // lane group (ISA 7.12.2 layouts)

    v8f acc[4][4];
#pragma unroll
    for (int mt = 0; mt < 4; ++mt)
#pragma unroll
        for (int nt = 0; nt < 4; ++nt)
            acc[mt][nt] = (v8f){0.f, 0.f, 0.f, 0.f, 0.f, 0.f, 0.f, 0.f};

    const _Float16* aRow[4];
#pragma unroll
    for (int mt = 0; mt < 4; ++mt)
        aRow[mt] = A + (size_t)(mBase + mt * 16 + l15) * K_DIM;
    const _Float16* bRow[4];
#pragma unroll
    for (int nt = 0; nt < 4; ++nt)
        bRow[nt] = B + (size_t)(nBase + nt * 16 + l15) * K_DIM;

    for (int k = 0; k < K_DIM; k += 32) {
        // Warm near caches for the K slab ~16 iterations ahead.
#pragma unroll
        for (int mt = 0; mt < 4; ++mt)
            __builtin_prefetch(aRow[mt] + k + 512, 0, 3);
#pragma unroll
        for (int nt = 0; nt < 4; ++nt)
            __builtin_prefetch(bRow[nt] + k + 512, 0, 3);

        // B fragment: lane holds col N = l15, K = k + hi*16 .. +15 (contiguous)
        v16h bf[4];
#pragma unroll
        for (int nt = 0; nt < 4; ++nt)
            bf[nt] = *(const v16h*)(bRow[nt] + k + hi * 16);

        // A fragment: lane holds row M = l15; halves 0..7 at K = k + hi*16,
        // halves 8..15 at K = k + hi*16 + 8 (two contiguous b128 loads)
        v16h af[4];
#pragma unroll
        for (int mt = 0; mt < 4; ++mt) {
            v8h lo8 = *(const v8h*)(aRow[mt] + k + hi * 16);
            v8h hi8 = *(const v8h*)(aRow[mt] + k + hi * 16 + 8);
            af[mt] = __builtin_shufflevector(lo8, hi8,
                        0, 1, 2, 3, 4, 5, 6, 7, 8, 9, 10, 11, 12, 13, 14, 15);
        }

#pragma unroll
        for (int mt = 0; mt < 4; ++mt)
#pragma unroll
            for (int nt = 0; nt < 4; ++nt)
                acc[mt][nt] = __builtin_amdgcn_wmma_f32_16x16x32_f16(
                    /*neg_a=*/false, af[mt],
                    /*neg_b=*/false, bf[nt],
                    /*c_mod=*/(short)0, acc[mt][nt],
                    /*reuse_a=*/false, /*reuse_b=*/false);
    }

    // Epilogue: C/D layout — VGPR r: lanes 0-15 -> (M=r, N=lane),
    // lanes 16-31 -> (M=r+8, N=lane-16). Apply f32 scale & bias.
#pragma unroll
    for (int nt = 0; nt < 4; ++nt) {
        const int col = nBase + nt * 16 + l15;
        const float sc = scale[col];
        const float bs = bias[col];
#pragma unroll
        for (int mt = 0; mt < 4; ++mt) {
#pragma unroll
            for (int r = 0; r < 8; ++r) {
                const int rowm = mBase + mt * 16 + hi * 8 + r;
                C[(size_t)rowm * N_DIM + col] = acc[mt][nt][r] * sc + bs;
            }
        }
    }
}

// ---------------------------------------------------------------------------
extern "C" void kernel_launch(void* const* d_in, const int* in_sizes, int n_in,
                              void* d_out, int out_size, void* d_ws, size_t ws_size,
                              hipStream_t stream)
{
    (void)in_sizes; (void)n_in; (void)out_size; (void)ws_size;

    const float* x    = (const float*)d_in[0];   // [4,2048,4096] f32
    const float* w    = (const float*)d_in[1];   // [4096,4096]   f32
    const float* bias = (const float*)d_in[2];   // [4096]        f32
    float*       out  = (float*)d_out;           // [4,2048,4096] f32

    // Workspace layout:
    //   signs : N*K f16  = 32 MB
    //   scale : N   f32  = 16 KB
    //   xh    : M*K f16  = 64 MB
    char* ws = (char*)d_ws;
    _Float16* signs = (_Float16*)ws;
    float*    scale = (float*)(ws + (size_t)N_DIM * K_DIM * sizeof(_Float16));
    _Float16* xh    = (_Float16*)(ws + (size_t)N_DIM * K_DIM * sizeof(_Float16)
                                     + 16384);

    // Phase 1: weight stats + sign matrix (one block per output row)
    prep_weight_kernel<<<N_DIM, 256, 0, stream>>>(w, signs, scale);

    // Phase 2: x -> f16 (8 elems/thread, exact coverage of M*K)
    const int convBlocks = (int)(((size_t)M_DIM * K_DIM / 8) / 256);
    conv_x_kernel<<<convBlocks, 256, 0, stream>>>(x, xh);

    // Phase 3: WMMA GEMM; grid (N/256, M/128)
    gemm_wmma_kernel<<<dim3(N_DIM / 256, M_DIM / 128), 256, 0, stream>>>(
        xh, signs, scale, bias, out);
}
